// Part_65712999629561
// MI455X (gfx1250) — matrix operand order repaired and checked by reading, not compiled, not run on hardware
//
#include <hip/hip_runtime.h>
#include <math.h>

#define EPSC 1e-8f
#define LNE  1e-5f
#define NEGC -1000000000.0f

typedef __attribute__((ext_vector_type(16))) __bf16 v16bf;
typedef __attribute__((ext_vector_type(8)))  float  v8f;
typedef __attribute__((ext_vector_type(4)))  int    v4i;

#if __has_builtin(__builtin_amdgcn_global_load_async_to_lds_b128) && \
    __has_builtin(__builtin_amdgcn_global_store_async_from_lds_b128) && \
    __has_builtin(__builtin_amdgcn_s_wait_asynccnt)
#define HAVE_ASYNC_LDS 1
typedef __attribute__((address_space(1))) v4i* gv4p;  // global int4*
typedef __attribute__((address_space(3))) v4i* lv4p;  // LDS int4*
#else
#define HAVE_ASYNC_LDS 0
#endif

__device__ __forceinline__ v8f wmma_bf16(v16bf a, v16bf b, v8f c) {
  return __builtin_amdgcn_wmma_f32_16x16x32_bf16(false, a, false, b, (short)0, c, false, false);
}

// ---- fragment loaders (bf16 16x16x32, contiguous k = 16h + e order, same in A and B) ----
// A 16x32: lane<16 -> M=lane, K = e + 16*(lane>=16).
__device__ __forceinline__ v16bf load_a(const __bf16* base, int ld, int r0, int c0) {
  const int lane = threadIdx.x & 31;
  const __bf16* row = base + (size_t)(r0 + (lane & 15)) * ld + c0;
  const int kb = (lane & 16);
  v16bf f;
#pragma unroll
  for (int e = 0; e < 16; ++e) f[e] = row[e + kb];
  return f;
}
// A with only K<8 valid (per-head q), zeros elsewhere
__device__ __forceinline__ v16bf load_a_h8(const __bf16* base, int ld, int r0, int c0) {
  const int lane = threadIdx.x & 31;
  const __bf16* row = base + (size_t)(r0 + (lane & 15)) * ld + c0;
  const int kb = (lane & 16);
  v16bf f;
#pragma unroll
  for (int e = 0; e < 16; ++e) {
    int k = e + kb;
    f[e] = (k < 8) ? row[k] : (__bf16)0.0f;
  }
  return f;
}
// B 32x16: lane holds column N=lane&15; K = e + 16*(lane>=16).
// Source row-major [N][K] (weights W[out][in], or token-major K-matrix).
__device__ __forceinline__ v16bf load_b(const __bf16* base, int ld, int n0, int c0) {
  const int lane = threadIdx.x & 31;
  const __bf16* row = base + (size_t)(n0 + (lane & 15)) * ld + c0;
  const int kb = (lane & 16);
  v16bf f;
#pragma unroll
  for (int e = 0; e < 16; ++e) f[e] = row[e + kb];
  return f;
}
// B with only K<8 valid (per-head k^T)
__device__ __forceinline__ v16bf load_b_h8(const __bf16* base, int ld, int n0, int c0) {
  const int lane = threadIdx.x & 31;
  const __bf16* row = base + (size_t)(n0 + (lane & 15)) * ld + c0;
  const int kb = (lane & 16);
  v16bf f;
#pragma unroll
  for (int e = 0; e < 16; ++e) {
    int k = e + kb;
    f[e] = (k < 8) ? row[k] : (__bf16)0.0f;
  }
  return f;
}
// B[k][n] = v[k][c0+n], only n<8 valid (per-head V columns)
__device__ __forceinline__ v16bf load_b_vcol(const __bf16* qkv, int ld, int k0, int c0) {
  const int lane = threadIdx.x & 31;
  const int n = lane & 15;
  const int kb = lane & 16;
  v16bf f;
#pragma unroll
  for (int e = 0; e < 16; ++e) {
    int k = k0 + e + kb;
    f[e] = (n < 8) ? qkv[(size_t)k * ld + c0 + n] : (__bf16)0.0f;
  }
  return f;
}

__device__ __forceinline__ float geluf(float x) {
  return 0.5f * x * (1.0f + erff(x * 0.70710678118f));
}
__device__ __forceinline__ void ln_row_to_bf(const float* src, int nv, __bf16* dst) {
  float m = 0.f;
  for (int c = 0; c < nv; ++c) m += src[c];
  m /= (float)nv;
  float v = 0.f;
  for (int c = 0; c < nv; ++c) { float d = src[c] - m; v += d * d; }
  v /= (float)nv;
  float s = rsqrtf(v + LNE);
  for (int c = 0; c < nv; ++c) dst[c] = (__bf16)((src[c] - m) * s);
}

// ============ Kernel 1: batchnorm statistics (per channel over N*P) ============
__global__ __launch_bounds__(256) void k_bnstats(const float* __restrict__ x,
                                                 const float* __restrict__ fm,
                                                 float* __restrict__ mu, float* __restrict__ var) {
  __shared__ float ss[256], sq[256];
  const int c = blockIdx.x, tid = threadIdx.x;
  float s = 0.f, q = 0.f;
  for (int i = tid; i < 2048 * 64; i += 256) {
    int n = i >> 6, p = i & 63;
    float v = (c < 60) ? x[((size_t)n * 60 + c) * 64 + p]
                       : fm[((size_t)n * 4 + (c - 60)) * 64 + p];
    s += v; q += v * v;
  }
  ss[tid] = s; sq[tid] = q;
  __syncthreads();
  for (int o = 128; o > 0; o >>= 1) {
    if (tid < o) { ss[tid] += ss[tid + o]; sq[tid] += sq[tid + o]; }
    __syncthreads();
  }
  if (tid == 0) {
    float m = ss[0] * (1.0f / 131072.0f);
    mu[c] = m;
    var[c] = sq[0] * (1.0f / 131072.0f) - m * m;
  }
}

// ============ Kernel 2: f32 -> bf16 weight repack for particle blocks ============
__global__ void k_wconv(const float* __restrict__ win, const float* __restrict__ wout,
                        const float* __restrict__ wf1, const float* __restrict__ wf2,
                        __bf16* __restrict__ dst) {
  int i = blockIdx.x * blockDim.x + threadIdx.x;
  if (i >= 32768) return;
  float v;
  if (i < 12288) v = win[i];
  else if (i < 16384) v = wout[i - 12288];
  else if (i < 24576) v = wf1[i - 16384];
  else v = wf2[i - 24576];
  dst[i] = (__bf16)v;
}

// ============ Kernel 3: BN apply + embedding MLP (64->64->32), token/thread ============
__global__ __launch_bounds__(256) void k_embed(const float* __restrict__ x, const float* __restrict__ fm,
                                               const float* __restrict__ mu, const float* __restrict__ var,
                                               const float* __restrict__ bnw, const float* __restrict__ bnb,
                                               const float* __restrict__ e1w, const float* __restrict__ e1b,
                                               const float* __restrict__ e2w, const float* __restrict__ e2b,
                                               float* __restrict__ h) {
  __shared__ float w1[64 * 64], w2[32 * 64];
  __shared__ float sb1[64], sb2[32], smu[64], siv[64], sbw[64], sbb[64];
  const int tid = threadIdx.x;
  for (int i = tid; i < 4096; i += 256) w1[i] = e1w[i];
  for (int i = tid; i < 2048; i += 256) w2[i] = e2w[i];
  if (tid < 64) {
    sb1[tid] = e1b[tid]; smu[tid] = mu[tid];
    siv[tid] = rsqrtf(var[tid] + LNE);
    sbw[tid] = bnw[tid]; sbb[tid] = bnb[tid];
  }
  if (tid < 32) sb2[tid] = e2b[tid];
  __syncthreads();
  const int idx = blockIdx.x * 256 + tid;  // token id, exactly 2048*64 tokens
  const int n = idx >> 6, p = idx & 63;
  float t0[64];
  for (int c = 0; c < 60; ++c) t0[c] = x[((size_t)n * 60 + c) * 64 + p];
  for (int c = 0; c < 4; ++c) t0[60 + c] = fm[((size_t)n * 4 + c) * 64 + p];
  for (int c = 0; c < 64; ++c) t0[c] = (t0[c] - smu[c]) * siv[c] * sbw[c] + sbb[c];
  // LN64
  float m = 0.f; for (int c = 0; c < 64; ++c) m += t0[c]; m *= (1.f / 64);
  float v = 0.f; for (int c = 0; c < 64; ++c) { float d = t0[c] - m; v += d * d; } v *= (1.f / 64);
  float s = rsqrtf(v + LNE);
  for (int c = 0; c < 64; ++c) t0[c] = (t0[c] - m) * s;
  float t1[64];
  for (int j = 0; j < 64; ++j) {
    float a = sb1[j];
    for (int c = 0; c < 64; ++c) a += t0[c] * w1[j * 64 + c];
    t1[j] = geluf(a);
  }
  m = 0.f; for (int c = 0; c < 64; ++c) m += t1[c]; m *= (1.f / 64);
  v = 0.f; for (int c = 0; c < 64; ++c) { float d = t1[c] - m; v += d * d; } v *= (1.f / 64);
  s = rsqrtf(v + LNE);
  for (int c = 0; c < 64; ++c) t1[c] = (t1[c] - m) * s;
  for (int j = 0; j < 32; ++j) {
    float a = sb2[j];
    for (int c = 0; c < 64; ++c) a += t1[c] * w2[j * 64 + c];
    h[(size_t)idx * 32 + j] = geluf(a);
  }
}

// ============ Kernel 4: jet pairwise features + conv1(4,s4) + conv2(2,s2,p1) -> bias[n][4][9][9] ============
__global__ __launch_bounds__(256) void k_jetbias(const float* __restrict__ fm,
                                                 const float* __restrict__ c1w, const float* __restrict__ c1b,
                                                 const float* __restrict__ c2w, const float* __restrict__ c2b,
                                                 float* __restrict__ bias) {
  __shared__ float f[4 * 64 * 64];    // 64KB pairwise features
  __shared__ float c1p[4 * 18 * 18];  // conv1 output, zero-padded ring
  __shared__ float pe[64], ppx[64], ppy[64], ppz[64], ppt[64], pyr[64], pph[64];
  __shared__ float w1s[256], w2s[64], b1s[4], b2s[4];
  const int n = blockIdx.x, tid = threadIdx.x;
  for (int i = tid; i < 256; i += 256) w1s[i] = c1w[i];
  if (tid < 64) w2s[tid] = c2w[tid];
  if (tid < 4) { b1s[tid] = c1b[tid]; b2s[tid] = c2b[tid]; }
  if (tid < 64) {
    float e  = fm[((size_t)n * 4 + 0) * 64 + tid];
    float px = fm[((size_t)n * 4 + 1) * 64 + tid];
    float py = fm[((size_t)n * 4 + 2) * 64 + tid];
    float pz = fm[((size_t)n * 4 + 3) * 64 + tid];
    pe[tid] = e; ppx[tid] = px; ppy[tid] = py; ppz[tid] = pz;
    ppt[tid] = sqrtf(px * px + py * py + EPSC);
    pyr[tid] = 0.5f * logf((e + pz + EPSC) / (e - pz + EPSC));
    float ph = fmodf(atan2f(py, px + EPSC), 6.283185307179586f);
    if (ph < 0.f) ph += 6.283185307179586f;
    pph[tid] = ph;
  }
  for (int i = tid; i < 4 * 18 * 18; i += 256) c1p[i] = 0.0f;
  __syncthreads();
  for (int idx = tid; idx < 16384; idx += 256) {
    int ch = idx >> 12, rem = idx & 4095, i = rem >> 6, j = rem & 63;
    float v = 0.f;
    if (i != j) {
      float dy = pyr[i] - pyr[j], dp = pph[i] - pph[j];
      float delta = sqrtf(fabsf(dy * dy + dp * dp));
      if (ch == 0) v = delta;
      else if (ch == 1) v = fminf(ppt[i], ppt[j]) * delta;
      else if (ch == 2) v = fminf(ppt[i], ppt[j]) / (ppt[i] + ppt[j] + EPSC);
      else {
        float sx = ppx[i] + ppx[j], sy = ppy[i] + ppy[j], sz = ppz[i] + ppz[j];
        v = pe[i] * pe[i] + pe[j] * pe[j] - (sx * sx + sy * sy + sz * sz);
      }
    }
    f[idx] = v;
  }
  __syncthreads();
  for (int o = tid; o < 1024; o += 256) {
    int oc = o >> 8, rem = o & 255, oy = rem >> 4, ox = rem & 15;
    float a = b1s[oc];
    for (int ic = 0; ic < 4; ++ic)
      for (int ky = 0; ky < 4; ++ky)
        for (int kx = 0; kx < 4; ++kx)
          a += f[ic * 4096 + (oy * 4 + ky) * 64 + ox * 4 + kx] * w1s[((oc * 4 + ic) * 4 + ky) * 4 + kx];
    c1p[oc * 324 + (oy + 1) * 18 + (ox + 1)] = a;
  }
  __syncthreads();
  for (int o = tid; o < 324; o += 256) {
    int oc = o / 81, rem = o % 81, oy = rem / 9, ox = rem % 9;
    float a = b2s[oc];
    for (int ic = 0; ic < 4; ++ic)
      for (int ky = 0; ky < 2; ++ky)
        for (int kx = 0; kx < 2; ++kx)
          a += c1p[ic * 324 + (oy * 2 + ky) * 18 + (ox * 2 + kx)] * w2s[((oc * 4 + ic) * 2 + ky) * 2 + kx];
    bias[(size_t)n * 324 + o] = a;
  }
}

// ============ Kernel 5: 4 fused particle transformer layers, block-per-jet, all-WMMA GEMMs ============
__global__ __launch_bounds__(256) void k_particle(float* __restrict__ hbuf,
                                                  const __bf16* __restrict__ wbf,
                                                  const float* __restrict__ in_b,  // [4][96]
                                                  const float* __restrict__ out_b, // [4][32]
                                                  const float* __restrict__ f1_b,  // [4][64]
                                                  const float* __restrict__ f2_b,  // [4][32]
                                                  const float* __restrict__ wres,  // [4][32]
                                                  const float* __restrict__ bias_cpe,
                                                  const int* __restrict__ jet) {
  __shared__ float  sh_h[64 * 32];   // residual/current h
  __shared__ __bf16 sh_x[64 * 64];   // bf16 A operand / attn output
  __shared__ __bf16 sh_qkv[64 * 96]; // q|k|v token-major
  __shared__ float  sh_f[64 * 64];   // f32 GEMM output scratch
  __shared__ __bf16 sh_a[64 * 64];   // softmaxed probabilities
  __shared__ float  sh_bias[4 * 81];

  const int n = blockIdx.x, tid = threadIdx.x;
  const int wave = tid >> 5, lane = tid & 31;
  const int njet = jet[2 * n + 1];

#if HAVE_ASYNC_LDS
  {  // async DMA of h tile (8KB = 512 x b128) + attention-bias tile (81 x b128) into LDS
    gv4p gsrc = (gv4p)(hbuf + (size_t)n * 2048);
    lv4p ldst = (lv4p)sh_h;
#pragma unroll
    for (int i = 0; i < 2; ++i)
      __builtin_amdgcn_global_load_async_to_lds_b128(gsrc + tid + i * 256,
                                                     ldst + tid + i * 256, 0, 0);
    gv4p gb = (gv4p)(bias_cpe + (size_t)n * 324);
    lv4p lb = (lv4p)sh_bias;
    if (tid < 81)
      __builtin_amdgcn_global_load_async_to_lds_b128(gb + tid, lb + tid, 0, 0);
    __builtin_amdgcn_s_wait_asynccnt(0);
  }
#else
  for (int i = tid; i < 2048; i += 256) sh_h[i] = hbuf[(size_t)n * 2048 + i];
  for (int i = tid; i < 324; i += 256) sh_bias[i] = bias_cpe[(size_t)n * 324 + i];
#endif
  __syncthreads();

  const __bf16* Win  = wbf;               // [4][96][32]
  const __bf16* Wout = wbf + 4 * 96 * 32; // [4][32][32]
  const __bf16* Wf1  = Wout + 4 * 32 * 32;// [4][64][32]
  const __bf16* Wf2  = Wf1 + 4 * 64 * 32; // [4][32][64]

  for (int L = 0; L < 4; ++L) {
    const __bf16* wi = Win + L * 96 * 32;
    const __bf16* wo = Wout + L * 32 * 32;
    const __bf16* w1 = Wf1 + L * 64 * 32;
    const __bf16* w2 = Wf2 + L * 32 * 64;
    const float* bi = in_b + L * 96;
    const float* bo = out_b + L * 32;
    const float* b1 = f1_b + L * 64;
    const float* b2 = f2_b + L * 32;
    const float* wr = wres + L * 32;
    if (L < 3 && lane == 0) {  // prefetch next layer weights (global_prefetch_b8)
      __builtin_prefetch(wi + 96 * 32, 0, 0);
      __builtin_prefetch(w1 + 64 * 32, 0, 0);
    }

    // --- LN(h) -> sh_x bf16 (cols 0..31) ---
    if (tid < 64) ln_row_to_bf(sh_h + tid * 32, 32, sh_x + tid * 64);
    __syncthreads();

    // --- QKV: (64x32)x(32x96), 24 WMMA tiles over 8 waves ---
    for (int j = 0; j < 3; ++j) {
      int t = wave * 3 + j;
      int mt = t & 3, nt = t >> 2;
      v16bf a = load_a(sh_x, 64, mt * 16, 0);
      v16bf b = load_b(wi, 32, nt * 16, 0);
      v8f acc = {};
      acc = wmma_bf16(a, b, acc);
      int col = nt * 16 + (lane & 15);
      int r0 = mt * 16 + ((lane & 16) >> 1);
      float bb = bi[col];
      float scl = (col < 32) ? 0.35355339059f : 1.0f;  // q *= D^-0.5
#pragma unroll
      for (int r = 0; r < 8; ++r)
        sh_qkv[(r0 + r) * 96 + col] = (__bf16)((acc[r] + bb) * scl);
    }
    __syncthreads();

    // --- attention, one head at a time ---
    for (int h = 0; h < 4; ++h) {
      // scores 64x64 (K=8 via zero-masked 16x16x32 WMMA), 16 tiles over 8 waves
      for (int j = 0; j < 2; ++j) {
        int t = wave * 2 + j;
        int mt = t & 3, nt = t >> 2;
        v16bf a = load_a_h8(sh_qkv, 96, mt * 16, h * 8);
        v16bf b = load_b_h8(sh_qkv, 96, nt * 16, 32 + h * 8);
        v8f acc = {};
        acc = wmma_bf16(a, b, acc);
        int s2 = nt * 16 + (lane & 15);
        int r0 = mt * 16 + ((lane & 16) >> 1);
#pragma unroll
        for (int r = 0; r < 8; ++r) {
          int l = r0 + r;
          float v = acc[r];
          if (l < 9 && s2 < 9) v += sh_bias[h * 81 + l * 9 + s2];
          if (s2 >= njet) v = NEGC;
          sh_f[l * 64 + s2] = v;
        }
      }
      __syncthreads();
      // softmax per row
      if (tid < 64) {
        const float* r = sh_f + tid * 64;
        float mx = r[0];
        for (int c = 1; c < 64; ++c) mx = fmaxf(mx, r[c]);
        float s = 0.f;
        for (int c = 0; c < 64; ++c) s += __expf(r[c] - mx);
        float inv = 1.0f / s;
        __bf16* o = sh_a + tid * 64;
        for (int c = 0; c < 64; ++c) o[c] = (__bf16)(__expf(r[c] - mx) * inv);
      }
      __syncthreads();
      // o_head = A(64x64) @ V_head(64x8), K=64 -> 2 WMMA k-steps, waves 0..3
      if (wave < 4) {
        v8f acc = {};
        for (int k0 = 0; k0 < 64; k0 += 32) {
          v16bf a = load_a(sh_a, 64, wave * 16, k0);
          v16bf b = load_b_vcol(sh_qkv, 96, k0, 64 + h * 8);
          acc = wmma_bf16(a, b, acc);
        }
        int nn = lane & 15;
        int r0 = wave * 16 + ((lane & 16) >> 1);
        if (nn < 8) {
#pragma unroll
          for (int r = 0; r < 8; ++r)
            sh_x[(r0 + r) * 64 + h * 8 + nn] = (__bf16)acc[r];
        }
      }
      __syncthreads();
    }

    // --- out proj: (64x32)x(32x32), 8 tiles ---
    {
      int mt = wave & 3, nt = wave >> 2;
      v16bf a = load_a(sh_x, 64, mt * 16, 0);
      v16bf b = load_b(wo, 32, nt * 16, 0);
      v8f acc = {};
      acc = wmma_bf16(a, b, acc);
      int col = nt * 16 + (lane & 15);
      int r0 = mt * 16 + ((lane & 16) >> 1);
      float bb = bo[col];
#pragma unroll
      for (int r = 0; r < 8; ++r) sh_f[(r0 + r) * 64 + col] = acc[r] + bb;
    }
    __syncthreads();
    // h = LN(attn_out) + res
    if (tid < 64) {
      const float* r = sh_f + tid * 64;
      float m = 0.f; for (int c = 0; c < 32; ++c) m += r[c]; m *= (1.f / 32);
      float v = 0.f; for (int c = 0; c < 32; ++c) { float d = r[c] - m; v += d * d; } v *= (1.f / 32);
      float s = rsqrtf(v + LNE);
      float* hh = sh_h + tid * 32;
      for (int c = 0; c < 32; ++c) hh[c] = (r[c] - m) * s + hh[c];
    }
    __syncthreads();

    // --- FFN ---
    if (tid < 64) ln_row_to_bf(sh_h + tid * 32, 32, sh_x + tid * 64);
    __syncthreads();
    for (int j = 0; j < 2; ++j) {  // FFN1: 64x64 out, 16 tiles
      int t = wave * 2 + j;
      int mt = t & 3, nt = t >> 2;
      v16bf a = load_a(sh_x, 64, mt * 16, 0);
      v16bf b = load_b(w1, 32, nt * 16, 0);
      v8f acc = {};
      acc = wmma_bf16(a, b, acc);
      int col = nt * 16 + (lane & 15);
      int r0 = mt * 16 + ((lane & 16) >> 1);
      float bb = b1[col];
#pragma unroll
      for (int r = 0; r < 8; ++r) sh_f[(r0 + r) * 64 + col] = geluf(acc[r] + bb);
    }
    __syncthreads();
    if (tid < 64) ln_row_to_bf(sh_f + tid * 64, 64, sh_x + tid * 64);
    __syncthreads();
    {  // FFN2: 64x32 out, K=64 -> 2 k-steps, 8 tiles; fused residual h = g2 + wres*res
      int mt = wave & 3, nt = wave >> 2;
      v8f acc = {};
      for (int k0 = 0; k0 < 64; k0 += 32) {
        v16bf a = load_a(sh_x, 64, mt * 16, k0);
        v16bf b = load_b(w2, 64, nt * 16, k0);
        acc = wmma_bf16(a, b, acc);
      }
      int col = nt * 16 + (lane & 15);
      int r0 = mt * 16 + ((lane & 16) >> 1);
      float bb = b2[col], wrc = wr[col];
#pragma unroll
      for (int r = 0; r < 8; ++r) {
        float* hp = sh_h + (r0 + r) * 32 + col;
        *hp = (acc[r] + bb) + wrc * (*hp);
      }
    }
    __syncthreads();
  }
#if HAVE_ASYNC_LDS
  {  // async DMA of final h tile back to HBM
    gv4p gdst = (gv4p)(hbuf + (size_t)n * 2048);
    lv4p lsrc = (lv4p)sh_h;
#pragma unroll
    for (int i = 0; i < 2; ++i)
      __builtin_amdgcn_global_store_async_from_lds_b128(gdst + tid + i * 256,
                                                        lsrc + tid + i * 256, 0, 0);
    __builtin_amdgcn_s_wait_asynccnt(0);
  }
#else
  for (int i = tid; i < 2048; i += 256) hbuf[(size_t)n * 2048 + i] = sh_h[i];
#endif
}

// ============ Kernel 6: class attention (2 layers) + MLP head, block-per-jet ============
__global__ __launch_bounds__(128) void k_cls(const float* __restrict__ hbuf, const int* __restrict__ jet,
                                             const float* __restrict__ in_w, const float* __restrict__ in_b,
                                             const float* __restrict__ out_w, const float* __restrict__ out_b,
                                             const float* __restrict__ f1_w, const float* __restrict__ f1_b,
                                             const float* __restrict__ f2_w, const float* __restrict__ f2_b,
                                             const float* __restrict__ wres, const float* __restrict__ cattn,
                                             const float* __restrict__ cls_tok,
                                             const float* __restrict__ nw, const float* __restrict__ nb,
                                             const float* __restrict__ m1w, const float* __restrict__ m1b,
                                             const float* __restrict__ mhw, const float* __restrict__ mhb,
                                             const float* __restrict__ mfw, const float* __restrict__ mfb,
                                             float* __restrict__ out) {
  __shared__ float uh[64 * 32], kb_[65 * 32], vb_[65 * 32];
  __shared__ float cls[32], clsres[32], u0[32], q[32], sc[4 * 65];
  __shared__ float oatt[32], aout[32], perm[32], gbuf[64], gl[64], z0[32], z1[32], z2[32];
  const int n = blockIdx.x, tid = threadIdx.x;
  const int njet = jet[2 * n + 1];
  if (tid < 64) {  // ln of h tokens, shared by both layers
    const float* r = hbuf + (size_t)n * 2048 + tid * 32;
    float m = 0.f; for (int c = 0; c < 32; ++c) m += r[c]; m *= (1.f / 32);
    float v = 0.f; for (int c = 0; c < 32; ++c) { float d = r[c] - m; v += d * d; } v *= (1.f / 32);
    float s = rsqrtf(v + LNE);
    for (int c = 0; c < 32; ++c) uh[tid * 32 + c] = (r[c] - m) * s;
  }
  if (tid < 32) cls[tid] = cls_tok[tid];
  __syncthreads();
  for (int L = 0; L < 2; ++L) {
    const float* wi = in_w + L * 96 * 32;
    const float* bi = in_b + L * 96;
    const float* wo = out_w + L * 32 * 32;
    const float* bo = out_b + L * 32;
    const float* w1 = f1_w + L * 64 * 32;
    const float* b1 = f1_b + L * 64;
    const float* w2 = f2_w + L * 32 * 64;
    const float* b2 = f2_b + L * 32;
    if (tid < 32) clsres[tid] = cls[tid];
    __syncthreads();
    if (tid == 0) {  // u0 = ln(cls)
      float m = 0.f; for (int c = 0; c < 32; ++c) m += cls[c]; m *= (1.f / 32);
      float v = 0.f; for (int c = 0; c < 32; ++c) { float d = cls[c] - m; v += d * d; } v *= (1.f / 32);
      float s = rsqrtf(v + LNE);
      for (int c = 0; c < 32; ++c) u0[c] = (cls[c] - m) * s;
    }
    __syncthreads();
    if (tid < 32) {  // q from raw cls
      float a = bi[tid];
      for (int c = 0; c < 32; ++c) a += cls[c] * wi[tid * 32 + c];
      q[tid] = a * 0.35355339059f;
    }
    for (int i2 = tid; i2 < 65 * 64; i2 += 128) {  // k,v projections over 65 tokens
      int t = i2 >> 6, j = i2 & 63;
      const float* src = (t == 0) ? u0 : (uh + (t - 1) * 32);
      int wrow = 32 + j;  // j<32 -> K rows 32..63, j>=32 -> V rows 64..95
      float a = bi[wrow];
      for (int c = 0; c < 32; ++c) a += src[c] * wi[wrow * 32 + c];
      if (j < 32) kb_[t * 32 + j] = a; else vb_[t * 32 + (j - 32)] = a;
    }
    __syncthreads();
    for (int i2 = tid; i2 < 260; i2 += 128) {  // scores
      int h = i2 / 65, t = i2 % 65;
      float s = 0.f;
      for (int d = 0; d < 8; ++d) s += q[h * 8 + d] * kb_[t * 32 + h * 8 + d];
      if (t > 0 && (t - 1) >= njet) s = NEGC;
      sc[h * 65 + t] = s;
    }
    __syncthreads();
    if (tid < 4) {  // softmax per head
      float mx = sc[tid * 65];
      for (int t = 1; t < 65; ++t) mx = fmaxf(mx, sc[tid * 65 + t]);
      float s = 0.f;
      for (int t = 0; t < 65; ++t) { float e = expf(sc[tid * 65 + t] - mx); sc[tid * 65 + t] = e; s += e; }
      float inv = 1.0f / s;
      for (int t = 0; t < 65; ++t) sc[tid * 65 + t] *= inv;
    }
    __syncthreads();
    if (tid < 32) {
      int h = tid >> 3, d = tid & 7;
      float a = 0.f;
      for (int t = 0; t < 65; ++t) a += sc[h * 65 + t] * vb_[t * 32 + h * 8 + d];
      oatt[tid] = a;
    }
    __syncthreads();
    if (tid < 32) {
      float a = bo[tid];
      for (int c = 0; c < 32; ++c) a += oatt[c] * wo[tid * 32 + c];
      aout[tid] = a;
    }
    __syncthreads();
    if (tid < 32) {  // einsum 'tbhd,h->tbdh'
      int d = tid >> 2, h = tid & 3;
      perm[tid] = aout[h * 8 + d] * cattn[L * 4 + h];
    }
    __syncthreads();
    if (tid == 0) {  // cls = ln(perm) + res
      float m = 0.f; for (int c = 0; c < 32; ++c) m += perm[c]; m *= (1.f / 32);
      float v = 0.f; for (int c = 0; c < 32; ++c) { float d = perm[c] - m; v += d * d; } v *= (1.f / 32);
      float s = rsqrtf(v + LNE);
      for (int c = 0; c < 32; ++c) cls[c] = (perm[c] - m) * s + clsres[c];
    }
    __syncthreads();
    if (tid < 32) clsres[tid] = cls[tid];
    __syncthreads();
    if (tid == 0) {  // u0 = ln(cls)
      float m = 0.f; for (int c = 0; c < 32; ++c) m += cls[c]; m *= (1.f / 32);
      float v = 0.f; for (int c = 0; c < 32; ++c) { float d = cls[c] - m; v += d * d; } v *= (1.f / 32);
      float s = rsqrtf(v + LNE);
      for (int c = 0; c < 32; ++c) u0[c] = (cls[c] - m) * s;
    }
    __syncthreads();
    if (tid < 64) {
      float a = b1[tid];
      for (int c = 0; c < 32; ++c) a += u0[c] * w1[tid * 32 + c];
      gbuf[tid] = geluf(a);
    }
    __syncthreads();
    if (tid == 0) {  // ln(g)
      float m = 0.f; for (int c = 0; c < 64; ++c) m += gbuf[c]; m *= (1.f / 64);
      float v = 0.f; for (int c = 0; c < 64; ++c) { float d = gbuf[c] - m; v += d * d; } v *= (1.f / 64);
      float s = rsqrtf(v + LNE);
      for (int c = 0; c < 64; ++c) gl[c] = (gbuf[c] - m) * s;
    }
    __syncthreads();
    if (tid < 32) {
      float a = b2[tid];
      for (int c = 0; c < 64; ++c) a += gl[c] * w2[tid * 64 + c];
      cls[tid] = a + wres[L * 32 + tid] * clsres[tid];
    }
    __syncthreads();
  }
  // head
  if (tid == 0) {
    float m = 0.f; for (int c = 0; c < 32; ++c) m += cls[c]; m *= (1.f / 32);
    float v = 0.f; for (int c = 0; c < 32; ++c) { float d = cls[c] - m; v += d * d; } v *= (1.f / 32);
    float s = rsqrtf(v + LNE);
    for (int c = 0; c < 32; ++c) z0[c] = (cls[c] - m) * s * nw[c] + nb[c];
  }
  __syncthreads();
  if (tid < 32) {
    float a = m1b[tid];
    for (int c = 0; c < 32; ++c) a += z0[c] * m1w[tid * 32 + c];
    z1[tid] = fmaxf(a, 0.0f);
  }
  __syncthreads();
  if (tid < 32) {
    float a = mhb[tid];
    for (int c = 0; c < 32; ++c) a += z1[c] * mhw[tid * 32 + c];
    z2[tid] = fmaxf(a, 0.0f);
  }
  __syncthreads();
  if (tid == 0) {
    float a = mfb[0];
    for (int c = 0; c < 32; ++c) a += z2[c] * mfw[c];
    out[n] = 1.0f / (1.0f + expf(-a));
  }
}

extern "C" void kernel_launch(void* const* d_in, const int* in_sizes, int n_in,
                              void* d_out, int out_size, void* d_ws, size_t ws_size,
                              hipStream_t stream) {
  (void)in_sizes; (void)n_in; (void)out_size; (void)ws_size;
  const float* x    = (const float*)d_in[0];
  const float* fm   = (const float*)d_in[1];
  const int*   jet  = (const int*)d_in[2];
  const float* bn_w = (const float*)d_in[3];
  const float* bn_b = (const float*)d_in[4];
  const float* e1w  = (const float*)d_in[5];
  const float* e1b  = (const float*)d_in[6];
  const float* e2w  = (const float*)d_in[7];
  const float* e2b  = (const float*)d_in[8];
  const float* c1w  = (const float*)d_in[9];
  const float* c1b  = (const float*)d_in[10];
  const float* c2w  = (const float*)d_in[11];
  const float* c2b  = (const float*)d_in[12];

  float*  ws_mu   = (float*)d_ws;
  float*  ws_var  = ws_mu + 64;
  __bf16* ws_wbf  = (__bf16*)((char*)d_ws + 512);
  float*  ws_bias = (float*)((char*)d_ws + 512 + 32768 * 2);
  float*  ws_h    = ws_bias + 2048 * 324;

  k_bnstats<<<64, 256, 0, stream>>>(x, fm, ws_mu, ws_var);
  k_wconv<<<128, 256, 0, stream>>>((const float*)d_in[13], (const float*)d_in[15],
                                   (const float*)d_in[17], (const float*)d_in[19], ws_wbf);
  k_embed<<<512, 256, 0, stream>>>(x, fm, ws_mu, ws_var, bn_w, bn_b, e1w, e1b, e2w, e2b, ws_h);
  k_jetbias<<<2048, 256, 0, stream>>>(fm, c1w, c1b, c2w, c2b, ws_bias);
  k_particle<<<2048, 256, 0, stream>>>(ws_h, ws_wbf,
                                       (const float*)d_in[14], (const float*)d_in[16],
                                       (const float*)d_in[18], (const float*)d_in[20],
                                       (const float*)d_in[21], ws_bias, jet);
  k_cls<<<2048, 128, 0, stream>>>(ws_h, jet,
                                  (const float*)d_in[22], (const float*)d_in[23],
                                  (const float*)d_in[24], (const float*)d_in[25],
                                  (const float*)d_in[26], (const float*)d_in[27],
                                  (const float*)d_in[28], (const float*)d_in[29],
                                  (const float*)d_in[30], (const float*)d_in[31],
                                  (const float*)d_in[32], (const float*)d_in[33],
                                  (const float*)d_in[34], (const float*)d_in[35],
                                  (const float*)d_in[36], (const float*)d_in[37],
                                  (const float*)d_in[38], (const float*)d_in[39],
                                  (const float*)d_in[40], (float*)d_out);
}
